// _bilstm_crf_80977313399113
// MI455X (gfx1250) — compile-verified, matrix operand order checked by reading, not run on hardware
//
#include <hip/hip_runtime.h>
#include <hip/hip_bf16.h>
#include <cstdint>

#define SEQ   4096
#define EMB   500
#define EMBP  512
#define HID   200
#define G4    800      // 4*HID per direction
#define NPROJ 1600     // both directions concatenated
#define NTAG  50
#define NTAGP 64
#define KLIN  400      // 2*HID
#define KLINP 512      // padded K for classifier GEMM (power-of-2 rows for TDM padding)
#define LDSB  520      // LDS panel row stride in halves (512 + 8) -> conflict-free frags
#define STARTT 48
#define ENDT   49
#define NEGV  (-10000.0f)

typedef __attribute__((ext_vector_type(16))) _Float16 v16h;
typedef __attribute__((ext_vector_type(8)))  _Float16 v8h;
typedef __attribute__((ext_vector_type(8)))  float    v8f;
typedef __attribute__((ext_vector_type(4)))  unsigned int v4u;
typedef __attribute__((ext_vector_type(8)))  int      v8i;
typedef __attribute__((ext_vector_type(4)))  int      v4i;

#if defined(__has_builtin)
#if __has_builtin(__builtin_amdgcn_tensor_load_to_lds)
#define HAVE_TDM 1
#endif
#endif

// ---- workspace layout (bytes, all 256B aligned) ----
#define OFF_X16   ((size_t)0)          // 4096*512*2  = 4,194,304
#define OFF_WT16  ((size_t)4194304)    // 1600*512*2  = 1,638,400
#define OFF_XP    ((size_t)5832704)    // 4096*1600*4 = 26,214,400
#define OFF_WHH16 ((size_t)32047104)   // 2*800*200*2 = 640,000
#define OFF_H16   ((size_t)32687104)   // 4096*512*2  = 4,194,304
#define OFF_LW16  ((size_t)36881408)   // 64*512*2    = 65,536
#define OFF_FEATS ((size_t)36946944)   // 4096*64*4   = 1,048,576
#define OFF_BP    ((size_t)37995520)   // 4096*50*4   = 819,200

// ---------------- TDM 2D tile load (D# per CDNA5 ISA) -----------
#ifdef HAVE_TDM
__device__ inline void tdm_load_2d(void* ldsDst, const void* gSrc,
                                   unsigned cols, unsigned rows, unsigned strideElems,
                                   unsigned elemCode /*0=1B,1=2B,2=4B*/,
                                   bool pad /*interval=256DW, amount=4DW*/) {
  unsigned long long ga = (unsigned long long)(uintptr_t)gSrc;
  unsigned la = (unsigned)(uintptr_t)ldsDst;          // LDS byte offset
  v4u g0;
  g0[0] = 1u;                                         // count=1, user mode, no gather
  g0[1] = la;                                         // lds_addr
  g0[2] = (unsigned)(ga & 0xFFFFFFFFu);               // global_addr[31:0]
  g0[3] = (unsigned)((ga >> 32) & 0x01FFFFFFu) | (2u << 30);  // addr[56:32] | type=2
  v8i g1;
  g1[0] = (int)((elemCode << 16) |
                (pad ? ((1u << 20) | (7u << 22) | (3u << 25)) : 0u)); // mask=0,size,pad
  g1[1] = (int)((cols & 0xFFFFu) << 16);              // tensor_dim0[15:0] -> bits63:48
  g1[2] = (int)(((cols >> 16) & 0xFFFFu) | ((rows & 0xFFFFu) << 16)); // dim0 hi | dim1 lo
  g1[3] = (int)(((rows >> 16) & 0xFFFFu) | ((cols & 0xFFFFu) << 16)); // dim1 hi | tile0
  g1[4] = (int)(rows & 0xFFFFu);                      // tile_dim1 | tile_dim2=0
  g1[5] = (int)strideElems;                           // tensor_dim0_stride[31:0]
  g1[6] = 0;                                          // stride0 hi | stride1 lo
  g1[7] = 0;
  v4i z4 = {0, 0, 0, 0};
#if __clang_major__ >= 23
  v8i z8 = {0, 0, 0, 0, 0, 0, 0, 0};
  __builtin_amdgcn_tensor_load_to_lds(g0, g1, z4, z4, z8, 0);
#else
  __builtin_amdgcn_tensor_load_to_lds(g0, g1, z4, z4, 0);
#endif
}
#endif

// ---------------- WMMA fragment helper (16x16x32 f16) ----------------
// A 16x32: lane l -> row (l&15); halves 0..7 = K base+0..7, halves 8..15 = K base+16..23,
// base = (l<16)?0:8.  B mirrors with N in place of M.  Rows must be 16B aligned.
__device__ inline v16h load_frag16(const _Float16* base, int ld, int rowBase, int kBase) {
  const int lane = threadIdx.x & 31;
  const int r    = rowBase + (lane & 15);
  const int kOff = (lane & 16) ? 8 : 0;
  const _Float16* p = base + (size_t)r * ld + kBase + kOff;
  v8h lo = *reinterpret_cast<const v8h*>(p);
  v8h hi = *reinterpret_cast<const v8h*>(p + 16);
  v16h out;
#pragma unroll
  for (int i = 0; i < 8; ++i) { out[i] = lo[i]; out[i + 8] = hi[i]; }
  return out;
}

__device__ inline float sigf(float x) { return 1.0f / (1.0f + __expf(-x)); }

// ---------------- prep kernels ----------------
__global__ void k_gather_x(const int* __restrict__ ids, const float* __restrict__ emb,
                           _Float16* __restrict__ X16) {
  const int m = blockIdx.x;
  const float* src = emb + (size_t)ids[m] * EMB;
  for (int k = threadIdx.x; k < EMBP; k += blockDim.x)
    X16[(size_t)m * EMBP + k] = (k < EMB) ? (_Float16)src[k] : (_Float16)0.0f;
}

__global__ void k_prep_wih(const float* __restrict__ wf, const float* __restrict__ wb,
                           _Float16* __restrict__ WT16) {
  const int n = blockIdx.x;  // 0..1599
  const float* src = (n < G4) ? (wf + (size_t)n * EMB) : (wb + (size_t)(n - G4) * EMB);
  for (int k = threadIdx.x; k < EMBP; k += blockDim.x)
    WT16[(size_t)n * EMBP + k] = (k < EMB) ? (_Float16)src[k] : (_Float16)0.0f;
}

__global__ void k_prep_misc(const float* __restrict__ whhf, const float* __restrict__ whhb,
                            const float* __restrict__ lw, _Float16* __restrict__ Whh16,
                            _Float16* __restrict__ LW16, _Float16* __restrict__ H16) {
  int idx = blockIdx.x * blockDim.x + threadIdx.x;
  if (idx < 2 * HID * G4) {              // w_hh f32 -> f16, layout [dir][j=800][k=200]
    const int d = idx / (G4 * HID);
    const int r = idx % (G4 * HID);
    Whh16[idx] = (_Float16)((d ? whhb : whhf)[r]);
    return;
  }
  idx -= 2 * HID * G4;
  if (idx < NTAGP * KLINP) {             // linear_w -> padded f16 [64][512]
    const int n = idx / KLINP, k = idx % KLINP;
    LW16[idx] = (n < NTAG && k < KLIN) ? (_Float16)lw[(size_t)n * KLIN + k]
                                       : (_Float16)0.0f;
    return;
  }
  idx -= NTAGP * KLINP;
  if (idx < SEQ * (KLINP - KLIN)) {      // zero H16 pad columns 400..511
    const int m = idx / (KLINP - KLIN), k = KLIN + (idx % (KLINP - KLIN));
    H16[(size_t)m * KLINP + k] = (_Float16)0.0f;
  }
}

// ---------------- GEMM 1: XP[4096x1600] = X @ [Wihf;Wihb]^T + biases ----------------
__global__ __launch_bounds__(128) void k_gemm_xproj(
    const _Float16* __restrict__ X16, const _Float16* __restrict__ WT16,
    const float* __restrict__ bihf, const float* __restrict__ bhhf,
    const float* __restrict__ bihb, const float* __restrict__ bhhb,
    float* __restrict__ XP) {
  __shared__ __align__(16) _Float16 bpanel[64 * LDSB];   // 66,560 B, padded rows
  const int wave = threadIdx.x >> 5;
  const int lane = threadIdx.x & 31;
  const int mBase  = blockIdx.y * 64 + wave * 16;
  const int nBase0 = blockIdx.x * 64;
#ifdef HAVE_TDM
  if (threadIdx.x < 32) {     // one wave issues the tensor DMA for the whole block
    tdm_load_2d(bpanel, WT16 + (size_t)nBase0 * EMBP, EMBP, 64, EMBP, 1, true);
    __builtin_amdgcn_s_wait_tensorcnt(0);
  }
#else
  for (int i = threadIdx.x; i < 64 * EMBP; i += blockDim.x) {
    const int r = i / EMBP, c = i % EMBP;
    bpanel[r * LDSB + c] = WT16[(size_t)(nBase0 + r) * EMBP + c];
  }
#endif
  __syncthreads();
  v8f acc[4] = {};
  for (int kb = 0; kb < EMBP; kb += 32) {
    v16h a = load_frag16(X16, EMBP, mBase, kb);
#pragma unroll
    for (int t = 0; t < 4; ++t) {
      v16h b = load_frag16(bpanel, LDSB, t * 16, kb);
      acc[t] = __builtin_amdgcn_wmma_f32_16x16x32_f16(
          false, a, false, b, (short)0, acc[t], false, false);
    }
  }
  const int mo = (lane & 16) ? 8 : 0;
#pragma unroll
  for (int t = 0; t < 4; ++t) {
    const int n = nBase0 + t * 16 + (lane & 15);
    const float bias = (n < G4) ? (bihf[n] + bhhf[n]) : (bihb[n - G4] + bhhb[n - G4]);
#pragma unroll
    for (int v = 0; v < 8; ++v)
      XP[(size_t)(mBase + mo + v) * NPROJ + n] = acc[t][v] + bias;
  }
}

// ---------------- LSTM recurrence (persistent, one block per direction) ----------------
// Weights live in LDS as f16 [j=800][k=200]; f32 accumulation; h broadcast from LDS.
__global__ __launch_bounds__(800) void k_lstm(const float* __restrict__ XP,
                                              const _Float16* __restrict__ Whh16,
                                              const float* __restrict__ h0,
                                              const float* __restrict__ c0,
                                              _Float16* __restrict__ H16) {
  __shared__ __align__(16) _Float16 wlds[G4 * HID];      // 320,000 B
  __shared__ __align__(16) float hs[HID];
  __shared__ float gs[G4];
  const int dir = blockIdx.x;
  const int j = threadIdx.x;  // 0..799
#ifdef HAVE_TDM
  if (threadIdx.x < 32) {
    tdm_load_2d(wlds, Whh16 + (size_t)dir * G4 * HID, HID, G4, HID, 1, false);
    __builtin_amdgcn_s_wait_tensorcnt(0);
  }
#else
  for (int i = j; i < G4 * HID; i += G4) wlds[i] = Whh16[(size_t)dir * G4 * HID + i];
#endif
  float c = 0.0f;
  if (j < HID) { hs[j] = h0[dir * HID + j]; c = c0[dir * HID + j]; }
  __syncthreads();
  const _Float16* wrow = wlds + (size_t)j * HID;  // 400 B, 16B-aligned
  for (int t = 0; t < SEQ; ++t) {
    const int row = dir ? (SEQ - 1 - t) : t;
    float g = XP[(size_t)row * NPROJ + dir * G4 + j];
    if (t + 1 < SEQ) {
      const int nrow = dir ? (SEQ - 2 - t) : (t + 1);
      __builtin_prefetch(&XP[(size_t)nrow * NPROJ + dir * G4 + j], 0, 3);
    }
#pragma unroll
    for (int kb = 0; kb < HID; kb += 8) {
      v8h w = *reinterpret_cast<const v8h*>(wrow + kb);
      const float4 hA = *reinterpret_cast<const float4*>(hs + kb);
      const float4 hB = *reinterpret_cast<const float4*>(hs + kb + 4);
      g = fmaf((float)w[0], hA.x, g);
      g = fmaf((float)w[1], hA.y, g);
      g = fmaf((float)w[2], hA.z, g);
      g = fmaf((float)w[3], hA.w, g);
      g = fmaf((float)w[4], hB.x, g);
      g = fmaf((float)w[5], hB.y, g);
      g = fmaf((float)w[6], hB.z, g);
      g = fmaf((float)w[7], hB.w, g);
    }
    gs[j] = g;
    __syncthreads();
    if (j < HID) {
      const float gi = gs[j], gf = gs[j + HID], gg = gs[j + 2 * HID], go = gs[j + 3 * HID];
      c = sigf(gf) * c + sigf(gi) * tanhf(gg);
      const float h = sigf(go) * tanhf(c);
      hs[j] = h;
      H16[(size_t)row * KLINP + dir * HID + j] = (_Float16)h;
    }
    __syncthreads();
  }
}

// ---------------- GEMM 2: feats[4096x64] = H @ linear_w^T + b ----------------
__global__ __launch_bounds__(128) void k_gemm_linear(
    const _Float16* __restrict__ H16, const _Float16* __restrict__ LW16,
    const float* __restrict__ lb, float* __restrict__ feats) {
  __shared__ __align__(16) _Float16 lwpanel[64 * LDSB];
  const int wave = threadIdx.x >> 5;
  const int lane = threadIdx.x & 31;
  const int mBase = blockIdx.x * 64 + wave * 16;
#ifdef HAVE_TDM
  if (threadIdx.x < 32) {
    tdm_load_2d(lwpanel, LW16, KLINP, 64, KLINP, 1, true);
    __builtin_amdgcn_s_wait_tensorcnt(0);
  }
#else
  for (int i = threadIdx.x; i < 64 * KLINP; i += blockDim.x) {
    const int r = i / KLINP, c = i % KLINP;
    lwpanel[r * LDSB + c] = LW16[(size_t)r * KLINP + c];
  }
#endif
  __syncthreads();
  v8f acc[4] = {};
  for (int kb = 0; kb < KLINP; kb += 32) {
    v16h a = load_frag16(H16, KLINP, mBase, kb);
#pragma unroll
    for (int t = 0; t < 4; ++t) {
      v16h b = load_frag16(lwpanel, LDSB, t * 16, kb);
      acc[t] = __builtin_amdgcn_wmma_f32_16x16x32_f16(
          false, a, false, b, (short)0, acc[t], false, false);
    }
  }
  const int mo = (lane & 16) ? 8 : 0;
#pragma unroll
  for (int t = 0; t < 4; ++t) {
    const int n = t * 16 + (lane & 15);
    const float bias = (n < NTAG) ? lb[n] : 0.0f;
#pragma unroll
    for (int v = 0; v < 8; ++v)
      feats[(size_t)(mBase + mo + v) * NTAGP + n] = acc[t][v] + bias;
  }
}

// ---------------- CRF: forward logsumexp + Viterbi + gold + backtrace ----------------
__global__ __launch_bounds__(64) void k_crf(const float* __restrict__ feats,
                                            const float* __restrict__ trans,
                                            const int* __restrict__ tags,
                                            int* __restrict__ bp,
                                            float* __restrict__ out) {
  __shared__ float tr[NTAG * NTAG];
  __shared__ float alpha[NTAGP], val[NTAGP], nalpha[NTAGP], nval[NTAGP];
  const int j = threadIdx.x;
  for (int i = j; i < NTAG * NTAG; i += 64) tr[i] = trans[i];
  alpha[j] = (j == STARTT) ? 0.0f : NEGV;
  val[j]   = (j == STARTT) ? 0.0f : NEGV;
  __syncthreads();
  for (int t = 0; t < SEQ; ++t) {
    if (j < NTAG) {
      const float f = feats[(size_t)t * NTAGP + j];
      float mx = -3.4e38f;
      for (int i = 0; i < NTAG; ++i) mx = fmaxf(mx, alpha[i] + tr[i * NTAG + j]);
      float se = 0.0f;
      for (int i = 0; i < NTAG; ++i) se += __expf(alpha[i] + tr[i * NTAG + j] - mx);
      nalpha[j] = mx + __logf(se) + f;
      float vm = -3.4e38f; int ai = 0;
      for (int i = 0; i < NTAG; ++i) {
        const float s = val[i] + tr[i * NTAG + j];
        if (s > vm) { vm = s; ai = i; }
      }
      nval[j] = vm + f;
      bp[t * NTAG + j] = ai;
    }
    __syncthreads();
    if (j < NTAG) { alpha[j] = nalpha[j]; val[j] = nval[j]; }
    __syncthreads();
  }
  if (j == 0) {
    float mx = -3.4e38f;
    for (int i = 0; i < NTAG; ++i) mx = fmaxf(mx, alpha[i] + tr[i * NTAG + ENDT]);
    float se = 0.0f;
    for (int i = 0; i < NTAG; ++i) se += __expf(alpha[i] + tr[i * NTAG + ENDT] - mx);
    const float fwd = mx + __logf(se);
    float bm = -3.4e38f; int best = 0;
    for (int i = 0; i < NTAG; ++i) {
      const float s = val[i] + tr[i * NTAG + ENDT];
      if (s > bm) { bm = s; best = i; }
    }
    float gold = 0.0f; int prev = STARTT;
    for (int t = 0; t < SEQ; ++t) {
      const int tg = tags[t];
      gold += tr[prev * NTAG + tg] + feats[(size_t)t * NTAGP + tg];
      prev = tg;
    }
    gold += tr[prev * NTAG + ENDT];
    out[SEQ] = fwd - gold;
    int cur = best;
    out[SEQ - 1] = (float)cur;
    for (int t = SEQ - 1; t >= 1; --t) {
      cur = bp[t * NTAG + cur];
      out[t - 1] = (float)cur;
    }
  }
}

extern "C" void kernel_launch(void* const* d_in, const int* in_sizes, int n_in,
                              void* d_out, int out_size, void* d_ws, size_t ws_size,
                              hipStream_t stream) {
  const int*   ids   = (const int*)d_in[0];
  const int*   tags  = (const int*)d_in[1];
  const float* emb   = (const float*)d_in[2];
  const float* wihf  = (const float*)d_in[3];
  const float* whhf  = (const float*)d_in[4];
  const float* bihf  = (const float*)d_in[5];
  const float* bhhf  = (const float*)d_in[6];
  const float* wihb  = (const float*)d_in[7];
  const float* whhb  = (const float*)d_in[8];
  const float* bihb  = (const float*)d_in[9];
  const float* bhhb  = (const float*)d_in[10];
  const float* lw    = (const float*)d_in[11];
  const float* lb    = (const float*)d_in[12];
  const float* trans = (const float*)d_in[13];
  const float* h0    = (const float*)d_in[14];
  const float* c0    = (const float*)d_in[15];

  char* ws = (char*)d_ws;
  _Float16* X16   = (_Float16*)(ws + OFF_X16);
  _Float16* WT16  = (_Float16*)(ws + OFF_WT16);
  float*    XP    = (float*)(ws + OFF_XP);
  _Float16* Whh16 = (_Float16*)(ws + OFF_WHH16);
  _Float16* H16   = (_Float16*)(ws + OFF_H16);
  _Float16* LW16  = (_Float16*)(ws + OFF_LW16);
  float*    feats = (float*)(ws + OFF_FEATS);
  int*      bp    = (int*)(ws + OFF_BP);

  k_gather_x<<<SEQ, 256, 0, stream>>>(ids, emb, X16);
  k_prep_wih<<<NPROJ, 256, 0, stream>>>(wihf, wihb, WT16);
  k_prep_misc<<<3170, 256, 0, stream>>>(whhf, whhb, lw, Whh16, LW16, H16);
  k_gemm_xproj<<<dim3(25, 64), 128, 0, stream>>>(X16, WT16, bihf, bhhf, bihb, bhhb, XP);
  k_lstm<<<2, 800, 0, stream>>>(XP, Whh16, h0, c0, H16);
  k_gemm_linear<<<64, 128, 0, stream>>>(H16, LW16, lb, feats);
  k_crf<<<1, 64, 0, stream>>>(feats, trans, tags, bp, (float*)d_out);
}
